// DeepseekV2SteeringAttention_64622077936224
// MI455X (gfx1250) — matrix operand order repaired and checked by reading, not compiled, not run on hardware
//
#include <hip/hip_runtime.h>
#include <hip/hip_bf16.h>
#include <cstdint>
#include <cstddef>

// ---- problem constants ----
#define B_    2
#define S_    2048
#define HID_  2048
#define NH_   16
#define QLR_  1536
#define KVLR_ 512
#define DN_   128
#define DR_   64
#define DV_   128
#define DQK_  192          // DN + DR
#define ROWS_ (B_*S_)      // 4096
#define SCALE_ 0.07216878364870323f   // 1/sqrt(192)

typedef __attribute__((ext_vector_type(16))) __bf16 v16bf;
typedef __attribute__((ext_vector_type(8)))  __bf16 v8bf;
typedef __attribute__((ext_vector_type(8)))  float  v8f;

__device__ inline v8f v8f_zero(){ v8f z = {0.f,0.f,0.f,0.f,0.f,0.f,0.f,0.f}; return z; }

__device__ inline v8f wmma_bf16(v16bf a, v16bf b, v8f c){
  // D = A*B + C, fp32 accumulate; emits v_wmma_f32_16x16x32_bf16
  return __builtin_amdgcn_wmma_f32_16x16x32_bf16(false, a, false, b, (short)0, c, false, false);
}

// A-fragment: lane holds row M=lane%16; half = lane/16 selects K sub-ranges:
//  elems 0..7  -> K = kb + 8*half + {0..7}
//  elems 8..15 -> K = kb + 16 + 8*half + {0..7}
__device__ inline v16bf load_afrag(const __bf16* row, int kb, int hf){
  const __bf16* p = row + kb + 8*hf;
  v8bf lo = *reinterpret_cast<const v8bf*>(p);
  v8bf hi = *reinterpret_cast<const v8bf*>(p + 16);
  v16bf r;
#pragma unroll
  for (int i=0;i<8;++i){ r[i]=lo[i]; r[i+8]=hi[i]; }
  return r;
}

// B-fragment: lane holds column N=lane%16; elems 0..7 -> K = kb + 8*half + {0..7},
// elems 8..15 -> K = kb + 16 + 8*half + {0..7}, from a row of the transposed (N x K, row-major) operand.
__device__ inline v16bf load_bfrag(const __bf16* row, int kb, int hf){
  return load_afrag(row, kb, hf);
}

// ---------------- elementwise fp32 -> bf16 ----------------
__global__ __launch_bounds__(256) void k_f32_to_bf16(const float* __restrict__ in,
                                                     __bf16* __restrict__ out, size_t n){
  size_t i = (size_t)blockIdx.x*blockDim.x + threadIdx.x;
  size_t st = (size_t)gridDim.x*blockDim.x;
  for (; i < n; i += st) out[i] = (__bf16)in[i];
}

// ---------------- transpose (K x N fp32) -> (N x K bf16) ----------------
__global__ __launch_bounds__(256) void k_transpose_bf16(const float* __restrict__ in,
                                                        __bf16* __restrict__ out, int K, int N){
  __shared__ float tile[32][33];
  const int kb = blockIdx.y*32, nb = blockIdx.x*32;
  const int tx = threadIdx.x & 31, ty = threadIdx.x >> 5;
#pragma unroll
  for (int i = ty; i < 32; i += 8) tile[i][tx] = in[(size_t)(kb+i)*N + nb + tx];
  __syncthreads();
#pragma unroll
  for (int i = ty; i < 32; i += 8) out[(size_t)(nb+i)*K + kb + tx] = (__bf16)tile[tx][i];
}

// ---------------- bf16 GEMM: C(MxN fp32) = A(MxK bf16) @ Bt(NxK bf16)^T ----
// block = 256 thr = 8 waves (4 x 2); block tile 256x64; wave tile 64x32
// (4x2 WMMA frags -> 12 b128 loads per 8 WMMAs).
// Requires: M % 256 == 0, N % 64 == 0, K % 128 == 0.
__global__ __launch_bounds__(256) void k_gemm_bf16(const __bf16* __restrict__ A,
                                                   const __bf16* __restrict__ Bt,
                                                   float* __restrict__ C,
                                                   int M, int N, int K){
  const int lane = threadIdx.x & 31, wid = threadIdx.x >> 5;
  const int hf = lane >> 4, ln = lane & 15;
  const int m0 = blockIdx.y*256 + (wid & 3)*64;
  const int n0 = blockIdx.x*64  + (wid >> 2)*32;
  (void)M;
  v8f acc[4][2];
#pragma unroll
  for (int i = 0; i < 4; ++i){ acc[i][0] = v8f_zero(); acc[i][1] = v8f_zero(); }
  const __bf16* ar[4];
#pragma unroll
  for (int i = 0; i < 4; ++i) ar[i] = A + (size_t)(m0 + 16*i + ln)*K;
  const __bf16* br[2];
  br[0] = Bt + (size_t)(n0 + ln)*K;
  br[1] = br[0] + (size_t)16*K;

  for (int kb = 0; kb < K; kb += 128){
    // pull the next 256B line of each row into the WGP cache (global_prefetch_b8)
    if (kb + 128 < K){
#pragma unroll
      for (int i = 0; i < 4; ++i) __builtin_prefetch(ar[i] + kb + 128, 0, 0);
#pragma unroll
      for (int i = 0; i < 2; ++i) __builtin_prefetch(br[i] + kb + 128, 0, 0);
    }
#pragma unroll
    for (int kk = 0; kk < 128; kk += 32){
      const int k = kb + kk;
      v16bf b0 = load_bfrag(br[0], k, hf);
      v16bf b1 = load_bfrag(br[1], k, hf);
#pragma unroll
      for (int i = 0; i < 4; ++i){
        v16bf a = load_afrag(ar[i], k, hf);
        acc[i][0] = wmma_bf16(a, b0, acc[i][0]);
        acc[i][1] = wmma_bf16(a, b1, acc[i][1]);
      }
    }
  }
#pragma unroll
  for (int mf = 0; mf < 4; ++mf)
#pragma unroll
    for (int nf = 0; nf < 2; ++nf)
#pragma unroll
      for (int r = 0; r < 8; ++r)
        C[(size_t)(m0 + mf*16 + r + 8*hf)*N + n0 + nf*16 + ln] = acc[mf][nf][r];
}

// ---------------- RMSNorm rows of fp32 -> bf16 ----------------
__global__ __launch_bounds__(256) void k_rmsnorm_bf16(const float* __restrict__ x,
                                                      const float* __restrict__ w,
                                                      __bf16* __restrict__ y,
                                                      int C, int inStride){
  const float* xr = x + (size_t)blockIdx.x*inStride;
  float ss = 0.f;
  for (int c = threadIdx.x; c < C; c += 256){ float v = xr[c]; ss += v*v; }
#pragma unroll
  for (int off = 16; off > 0; off >>= 1) ss += __shfl_xor(ss, off, 32);
  __shared__ float red[8];
  __shared__ float sscale;
  const int lane = threadIdx.x & 31, wid = threadIdx.x >> 5;
  if (lane == 0) red[wid] = ss;
  __syncthreads();
  if (threadIdx.x == 0){
    float t = 0.f;
#pragma unroll
    for (int i = 0; i < 8; ++i) t += red[i];
    sscale = rsqrtf(t/(float)C + 1e-6f);
  }
  __syncthreads();
  const float sc = sscale;
  __bf16* yr = y + (size_t)blockIdx.x*C;
  for (int c = threadIdx.x; c < C; c += 256) yr[c] = (__bf16)(xr[c]*sc*w[c]);
}

// ---------------- build Q,K (with RoPE) and V^T in bf16 ----------------
// q:  (ROWS, NH*DQK) fp32, kv: (ROWS, NH*(DN+DV)) fp32, ckv: (ROWS, 576) fp32
// Qb/Kb: (B,NH,S,DQK) bf16; Vt: (B,NH,DV,S) bf16
__global__ __launch_bounds__(256) void k_prep_qkv(const float* __restrict__ q,
                                                  const float* __restrict__ kv,
                                                  const float* __restrict__ ckv,
                                                  const float* __restrict__ cosp,
                                                  const float* __restrict__ sinp,
                                                  __bf16* __restrict__ Qb,
                                                  __bf16* __restrict__ Kb,
                                                  __bf16* __restrict__ Vt){
  const int row = blockIdx.x;
  const int b = row / S_, s = row % S_;
  const float* qr  = q   + (size_t)row*(NH_*DQK_);
  const float* kvr = kv  + (size_t)row*(NH_*(DN_+DV_));
  const float* kpe = ckv + (size_t)row*(KVLR_+DR_) + KVLR_;
  const float* cs = cosp + (size_t)s*DR_;
  const float* sn = sinp + (size_t)s*DR_;
  for (int idx = threadIdx.x; idx < NH_*DQK_; idx += 256){
    int h = idx / DQK_, d = idx % DQK_;
    float v;
    if (d < DN_) v = qr[h*DQK_ + d];
    else {
      int d2 = d - DN_;
      float x   = qr[h*DQK_ + DN_ + d2];
      float rot = (d2 < DR_/2) ? -qr[h*DQK_ + DN_ + d2 + DR_/2]
                               :  qr[h*DQK_ + DN_ + d2 - DR_/2];
      v = x*cs[d2] + rot*sn[d2];
    }
    Qb[((size_t)(b*NH_ + h)*S_ + s)*DQK_ + d] = (__bf16)v;
  }
  for (int idx = threadIdx.x; idx < NH_*DQK_; idx += 256){
    int h = idx / DQK_, d = idx % DQK_;
    float v;
    if (d < DN_) v = kvr[h*(DN_+DV_) + d];
    else {
      int d2 = d - DN_;
      float x   = kpe[d2];
      float rot = (d2 < DR_/2) ? -kpe[d2 + DR_/2] : kpe[d2 - DR_/2];
      v = x*cs[d2] + rot*sn[d2];
    }
    Kb[((size_t)(b*NH_ + h)*S_ + s)*DQK_ + d] = (__bf16)v;
  }
  for (int idx = threadIdx.x; idx < NH_*DV_; idx += 256){
    int h = idx / DV_, dv = idx % DV_;
    Vt[((size_t)(b*NH_ + h)*DV_ + dv)*S_ + s] = (__bf16)kvr[h*(DN_+DV_) + DN_ + dv];
  }
}

// ---------------- flash attention (causal) with WMMA ----------------
// grid (S/128, NH, B); 8 waves; wave handles 16 query rows; key blocks of 128.
__global__ __launch_bounds__(256) void k_mla_attention(const __bf16* __restrict__ Qb,
                                                       const __bf16* __restrict__ Kb,
                                                       const __bf16* __restrict__ Vt,
                                                       __bf16* __restrict__ attn){
  __shared__ __bf16 pls[8][16][128];   // per-wave P tile (C-layout -> A-layout bounce), 32 KB
  const int qt = blockIdx.x, h = blockIdx.y, b = blockIdx.z;
  const int lane = threadIdx.x & 31, w = threadIdx.x >> 5;
  const int hf = lane >> 4, ln = lane & 15;
  const size_t bh = (size_t)(b*NH_ + h);
  const __bf16* Qp = Qb + bh*(size_t)S_*DQK_;
  const __bf16* Kp = Kb + bh*(size_t)S_*DQK_;
  const __bf16* Vp = Vt + bh*(size_t)DV_*S_;
  const int q0 = qt*128 + w*16;

  v16bf aq[6];
  const __bf16* qrow = Qp + (size_t)(q0 + ln)*DQK_;
#pragma unroll
  for (int ks = 0; ks < 6; ++ks) aq[ks] = load_afrag(qrow, ks*32, hf);

  v8f o[8];
  float mi[8], li[8];
#pragma unroll
  for (int i = 0; i < 8; ++i){ o[i] = v8f_zero(); mi[i] = -1e30f; li[i] = 0.f; }

  for (int jt = 0; jt <= qt; ++jt){
    // ---- S = Q @ K^T  (16 x 128 per wave) ----
    v8f sacc[8];
#pragma unroll
    for (int i = 0; i < 8; ++i) sacc[i] = v8f_zero();
#pragma unroll
    for (int ks = 0; ks < 6; ++ks){
#pragma unroll
      for (int nt = 0; nt < 8; ++nt){
        const __bf16* krow = Kp + (size_t)(jt*128 + nt*16 + ln)*DQK_;
        v16bf bk = load_bfrag(krow, ks*32, hf);
        sacc[nt] = wmma_bf16(aq[ks], bk, sacc[nt]);
      }
    }
    // ---- scale + causal mask ----
    const bool diag = (jt == qt);
#pragma unroll
    for (int nt = 0; nt < 8; ++nt)
#pragma unroll
      for (int r = 0; r < 8; ++r){
        float v = sacc[nt][r] * SCALE_;
        if (diag){
          int key = jt*128 + nt*16 + ln;
          int qry = q0 + r + 8*hf;
          if (key > qry) v = -1e30f;
        }
        sacc[nt][r] = v;
      }
    // ---- online softmax (row = r + 8*hf; row spans 16 lanes of the half) ----
#pragma unroll
    for (int r = 0; r < 8; ++r){
      float mx = sacc[0][r];
#pragma unroll
      for (int nt = 1; nt < 8; ++nt) mx = fmaxf(mx, sacc[nt][r]);
#pragma unroll
      for (int off = 1; off <= 8; off <<= 1) mx = fmaxf(mx, __shfl_xor(mx, off, 32));
      float mnew = fmaxf(mi[r], mx);
      float corr = __expf(mi[r] - mnew);
      mi[r] = mnew;
      float rs = 0.f;
#pragma unroll
      for (int nt = 0; nt < 8; ++nt){
        float p = __expf(sacc[nt][r] - mnew);
        sacc[nt][r] = p; rs += p;
      }
#pragma unroll
      for (int off = 1; off <= 8; off <<= 1) rs += __shfl_xor(rs, off, 32);
      li[r] = li[r]*corr + rs;
#pragma unroll
      for (int nt = 0; nt < 8; ++nt) o[nt][r] = o[nt][r]*corr;
    }
    // ---- P: C-layout -> LDS -> A-layout ----
#pragma unroll
    for (int nt = 0; nt < 8; ++nt)
#pragma unroll
      for (int r = 0; r < 8; ++r)
        pls[w][r + 8*hf][nt*16 + ln] = (__bf16)sacc[nt][r];
    __syncthreads();
    // ---- O += P @ V  (V^T rows are contiguous in key index) ----
#pragma unroll
    for (int kc = 0; kc < 4; ++kc){
      v16bf ap = load_afrag(&pls[w][ln][0], kc*32, hf);
#pragma unroll
      for (int nt = 0; nt < 8; ++nt){
        const __bf16* vrow = Vp + (size_t)(nt*16 + ln)*S_ + jt*128;
        v16bf bv = load_bfrag(vrow, kc*32, hf);
        o[nt] = wmma_bf16(ap, bv, o[nt]);
      }
    }
    __syncthreads();
  }
  // ---- normalize and store bf16 into (B*S, NH*DV) for the output GEMM ----
#pragma unroll
  for (int r = 0; r < 8; ++r){
    float inv = 1.0f / li[r];
    size_t grow = (size_t)(b*S_ + q0 + r + 8*hf);
#pragma unroll
    for (int nt = 0; nt < 8; ++nt)
      attn[grow*(NH_*DV_) + h*DV_ + nt*16 + ln] = (__bf16)(o[nt][r]*inv);
  }
}

// ======================= host orchestration =======================
extern "C" void kernel_launch(void* const* d_in, const int* in_sizes, int n_in,
                              void* d_out, int out_size, void* d_ws, size_t ws_size,
                              hipStream_t stream) {
  (void)in_sizes; (void)n_in; (void)out_size; (void)ws_size;
  const float* hs     = (const float*)d_in[0];
  /* d_in[1] = attention_mask, implemented analytically as causal */
  const float* cosp   = (const float*)d_in[2];
  const float* sinp   = (const float*)d_in[3];
  const float* w_qa   = (const float*)d_in[4];
  const float* qa_ln  = (const float*)d_in[5];
  const float* w_qb   = (const float*)d_in[6];
  const float* w_kva  = (const float*)d_in[7];
  const float* kva_ln = (const float*)d_in[8];
  const float* w_kvb  = (const float*)d_in[9];
  const float* w_o    = (const float*)d_in[10];
  float* out = (float*)d_out;

  char* ws = (char*)d_ws;
  size_t off = 0;
  auto alloc = [&](size_t bytes) -> char* {
    char* p = ws + off;
    off += (bytes + 255) & ~(size_t)255;
    return p;
  };
  __bf16* hs_b   = (__bf16*)alloc((size_t)ROWS_*HID_*2);
  __bf16* wqaT   = (__bf16*)alloc((size_t)QLR_*HID_*2);
  __bf16* wqbT   = (__bf16*)alloc((size_t)(NH_*DQK_)*QLR_*2);
  __bf16* wkvaT  = (__bf16*)alloc((size_t)(KVLR_+DR_)*HID_*2);
  __bf16* wkvbT  = (__bf16*)alloc((size_t)(NH_*(DN_+DV_))*KVLR_*2);
  __bf16* woT    = (__bf16*)alloc((size_t)HID_*(NH_*DV_)*2);
  float*  qa     = (float*) alloc((size_t)ROWS_*QLR_*4);
  __bf16* qan    = (__bf16*)alloc((size_t)ROWS_*QLR_*2);
  float*  qf     = (float*) alloc((size_t)ROWS_*(NH_*DQK_)*4);
  float*  ckv    = (float*) alloc((size_t)ROWS_*(KVLR_+DR_)*4);
  __bf16* kcn    = (__bf16*)alloc((size_t)ROWS_*KVLR_*2);
  float*  kvf    = (float*) alloc((size_t)ROWS_*(NH_*(DN_+DV_))*4);
  __bf16* Qb     = (__bf16*)alloc((size_t)B_*NH_*S_*DQK_*2);
  __bf16* Kb     = (__bf16*)alloc((size_t)B_*NH_*S_*DQK_*2);
  __bf16* Vt     = (__bf16*)alloc((size_t)B_*NH_*DV_*S_*2);
  __bf16* attnfl = (__bf16*)alloc((size_t)ROWS_*(NH_*DV_)*2);

  // 1) activations + weights -> bf16 (weights transposed to N x K)
  k_f32_to_bf16<<<4096, 256, 0, stream>>>(hs, hs_b, (size_t)ROWS_*HID_);
  k_transpose_bf16<<<dim3(QLR_/32,  HID_/32),  256, 0, stream>>>(w_qa,  wqaT,  HID_,  QLR_);
  k_transpose_bf16<<<dim3((NH_*DQK_)/32, QLR_/32), 256, 0, stream>>>(w_qb, wqbT, QLR_, NH_*DQK_);
  k_transpose_bf16<<<dim3((KVLR_+DR_)/32, HID_/32), 256, 0, stream>>>(w_kva, wkvaT, HID_, KVLR_+DR_);
  k_transpose_bf16<<<dim3((NH_*(DN_+DV_))/32, KVLR_/32), 256, 0, stream>>>(w_kvb, wkvbT, KVLR_, NH_*(DN_+DV_));
  k_transpose_bf16<<<dim3(HID_/32, (NH_*DV_)/32), 256, 0, stream>>>(w_o, woT, NH_*DV_, HID_);

  // 2) q_a = hs @ w_qa ; rmsnorm ; q = . @ w_qb
  k_gemm_bf16<<<dim3(QLR_/64, ROWS_/256), 256, 0, stream>>>(hs_b, wqaT, qa, ROWS_, QLR_, HID_);
  k_rmsnorm_bf16<<<ROWS_, 256, 0, stream>>>(qa, qa_ln, qan, QLR_, QLR_);
  k_gemm_bf16<<<dim3((NH_*DQK_)/64, ROWS_/256), 256, 0, stream>>>(qan, wqbT, qf, ROWS_, NH_*DQK_, QLR_);

  // 3) ckv = hs @ w_kva ; rmsnorm(k_c) ; kv = . @ w_kvb
  k_gemm_bf16<<<dim3((KVLR_+DR_)/64, ROWS_/256), 256, 0, stream>>>(hs_b, wkvaT, ckv, ROWS_, KVLR_+DR_, HID_);
  k_rmsnorm_bf16<<<ROWS_, 256, 0, stream>>>(ckv, kva_ln, kcn, KVLR_, KVLR_+DR_);
  k_gemm_bf16<<<dim3((NH_*(DN_+DV_))/64, ROWS_/256), 256, 0, stream>>>(kcn, wkvbT, kvf, ROWS_, NH_*(DN_+DV_), KVLR_);

  // 4) RoPE + pack Q/K/V^T in bf16
  k_prep_qkv<<<ROWS_, 256, 0, stream>>>(qf, kvf, ckv, cosp, sinp, Qb, Kb, Vt);

  // 5) causal flash attention with WMMA
  k_mla_attention<<<dim3(S_/128, NH_, B_), 256, 0, stream>>>(Qb, Kb, Vt, attnfl);

  // 6) output projection -> fp32 d_out
  k_gemm_bf16<<<dim3(HID_/64, ROWS_/256), 256, 0, stream>>>(attnfl, woT, out, ROWS_, HID_, NH_*DV_);
}
